// SRUpp__3856880631855
// MI455X (gfx1250) — matrix operand order repaired and checked by reading, not compiled, not run on hardware
//
#include <hip/hip_runtime.h>
#include <stdint.h>
#include <stddef.h>

// ---- problem dims (fixed by reference setup_inputs) ----
#define BB 4
#define LL 2048
#define DD 1024
#define NLAYERS 2
#define PP 256
#define HH 4
#define HD 64
#define MTOT (BB*LL)       // 8192 rows
#define EPSF 1e-5f

typedef __attribute__((ext_vector_type(16))) __bf16 v16bf;
typedef __attribute__((ext_vector_type(8)))  __bf16 v8bf;
typedef __attribute__((ext_vector_type(4)))  __bf16 v4bf;
typedef __attribute__((ext_vector_type(8)))  float  v8f;

// ---------------- fragment helpers (CDNA5 WMMA 16x16x32 bf16 layouts) -------
static __device__ inline v16bf cat8(v8bf lo, v8bf hi) {
  v16bf r;
#pragma unroll
  for (int j = 0; j < 8; ++j) { r[j] = lo[j]; r[j + 8] = hi[j]; }
  return r;
}

// A 16x32: lanes 0-15 hold M=lane, K = 0..15 ; lanes16-31 K = 16..31
static __device__ inline v16bf load_frag_a(const __bf16* base, int ld, int lane) {
  int m = lane & 15, hf = lane >> 4;
  const __bf16* r = base + (size_t)m * ld + hf * 16;
  v8bf lo = *(const v8bf*)(r);
  v8bf hi = *(const v8bf*)(r + 8);
  return cat8(lo, hi);
}

// B 32x16 where the contraction runs contiguously for each output column n
// (source rows = columns of B, leading-dim ld): lanes0-15 K=0..15, lanes16-31 K=16..31
static __device__ inline v16bf load_frag_b_rows(const __bf16* base, int ld, int lane) {
  int n = lane & 15, hf = lane >> 4;
  const __bf16* p = base + (size_t)n * ld + hf * 16;
  v8bf lo = *(const v8bf*)(p);
  v8bf hi = *(const v8bf*)(p + 8);
  return cat8(lo, hi);
}

// B 32x16 from a row-major [K][N] tile (strided gather per lane)
static __device__ inline v16bf load_frag_b_strided(const __bf16* base, int ld, int lane) {
  int n = lane & 15, hf = lane >> 4;
  const __bf16* p = base + (size_t)(hf * 16) * ld + n;
  v16bf r;
#pragma unroll
  for (int j = 0; j < 16; ++j) r[j] = p[(size_t)j * ld];
  return r;
}

static __device__ inline v8f wmma_bf16(v16bf a, v16bf b, v8f c) {
  return __builtin_amdgcn_wmma_f32_16x16x32_bf16(false, a, false, b, (short)0, c,
                                                 false, false);
}

// ---------------- CDNA5 async global->LDS copy ------------------------------
static __device__ inline void async_b128(uint32_t ldsoff, const void* g) {
  asm volatile("global_load_async_to_lds_b128 %0, %1, off"
               :: "v"(ldsoff), "v"(g) : "memory");
}
static __device__ inline void wait_async0() {
  asm volatile("s_wait_asynccnt 0" ::: "memory");
}

// ---------------- fp32 -> bf16 convert (vectorized) -------------------------
__global__ void k_cvt_bf16(const float* __restrict__ s, __bf16* __restrict__ d, int n) {
  int i = (blockIdx.x * blockDim.x + threadIdx.x) * 4;
  int stride = gridDim.x * blockDim.x * 4;
  for (; i < n; i += stride) {
    float4 f = *(const float4*)(s + i);
    v4bf o; o[0] = (__bf16)f.x; o[1] = (__bf16)f.y; o[2] = (__bf16)f.z; o[3] = (__bf16)f.w;
    *(v4bf*)(d + i) = o;
  }
}

// ---------------- fp32 [K][N] -> bf16 [N][K] transpose-convert --------------
// block 256 = 32x8, tile 32x32 through padded LDS (coalesced load & store)
__global__ __launch_bounds__(256)
void k_cvt_tr_bf16(const float* __restrict__ s, __bf16* __restrict__ d, int K, int N) {
  __shared__ float tile[32][33];
  const int tx = threadIdx.x & 31, ty = threadIdx.x >> 5;
  const int kb = blockIdx.y * 32, nb = blockIdx.x * 32;
#pragma unroll
  for (int j = 0; j < 32; j += 8)
    tile[ty + j][tx] = s[(size_t)(kb + ty + j) * N + nb + tx];
  __syncthreads();
#pragma unroll
  for (int j = 0; j < 32; j += 8)
    d[(size_t)(nb + ty + j) * K + kb + tx] = (__bf16)tile[tx][ty + j];
}

// ---------------- bf16 WMMA GEMM: C = A * Bt^T ------------------------------
// A: MxK row-major bf16. Bt: NxK row-major bf16 (pre-transposed weights).
// Block = 4 waves; each wave computes a 16(M) x 64(N) strip (4 accumulators,
// A fragment reused 4x). K-loop is software-pipelined with two fragment sets
// so loads for step k+32 are in flight while the WMMAs for step k execute.
// Grid: (N/64, M/64). No LDS, no barriers.
__global__ __launch_bounds__(128)
void k_gemm_bf16(const __bf16* __restrict__ A, const __bf16* __restrict__ Bt,
                 float* __restrict__ Cf, __bf16* __restrict__ Cb,
                 int M, int N, int K) {
  const int lane = threadIdx.x & 31;
  const int wave = threadIdx.x >> 5;
  const int n0 = blockIdx.x * 64;
  const int row0 = blockIdx.y * 64 + wave * 16;

  v8f zero = {};
  v8f acc[4];
#pragma unroll
  for (int t = 0; t < 4; ++t) acc[t] = zero;

  const __bf16* arow = A + (size_t)row0 * K;
  const __bf16* b0p = Bt + (size_t)n0 * K;

  // prologue: fragment set 0 at k=0
  v16bf a0 = load_frag_a(arow, K, lane);
  v16bf b0[4];
#pragma unroll
  for (int t = 0; t < 4; ++t) b0[t] = load_frag_b_rows(b0p + (size_t)(t * 16) * K, K, lane);

  for (int k0 = 0; k0 < K; k0 += 64) {    // K is a multiple of 64 (1024 or 256)
    // issue loads for set 1 (k0+32) before computing set 0
    v16bf a1 = load_frag_a(arow + k0 + 32, K, lane);
    v16bf b1[4];
#pragma unroll
    for (int t = 0; t < 4; ++t)
      b1[t] = load_frag_b_rows(b0p + (size_t)(t * 16) * K + k0 + 32, K, lane);
    __builtin_prefetch(arow + (size_t)k0 + 64, 0, 1);
#pragma unroll
    for (int t = 0; t < 4; ++t) acc[t] = wmma_bf16(a0, b0[t], acc[t]);

    // reload set 0 for k0+64 while set 1 computes
    if (k0 + 64 < K) {
      a0 = load_frag_a(arow + k0 + 64, K, lane);
#pragma unroll
      for (int t = 0; t < 4; ++t)
        b0[t] = load_frag_b_rows(b0p + (size_t)(t * 16) * K + k0 + 64, K, lane);
    }
#pragma unroll
    for (int t = 0; t < 4; ++t) acc[t] = wmma_bf16(a1, b1[t], acc[t]);
  }

  const int hf = lane >> 4;
#pragma unroll
  for (int i = 0; i < 8; ++i) {
    int row = row0 + i + 8 * hf;            // C/D layout: VGPR i -> M = i + 8*half
    size_t base = (size_t)row * N + n0 + (lane & 15);
#pragma unroll
    for (int t = 0; t < 4; ++t) {
      if (Cf) Cf[base + t * 16] = acc[t][i];
      if (Cb) Cb[base + t * 16] = (__bf16)acc[t][i];
    }
  }
}

// ---------------- LayerNorm over p=256, one wave per row --------------------
__global__ __launch_bounds__(256)
void k_layernorm(const float* __restrict__ z, const float* __restrict__ g,
                 const float* __restrict__ b, __bf16* __restrict__ out) {
  const int lane = threadIdx.x & 31;
  const int wave = threadIdx.x >> 5;
  const int row = blockIdx.x * 8 + wave;
  const float* zr = z + (size_t)row * PP;
  float v[8];
  float s = 0.f;
#pragma unroll
  for (int j = 0; j < 8; ++j) { v[j] = zr[lane * 8 + j]; s += v[j]; }
#pragma unroll
  for (int m = 1; m < 32; m <<= 1) s += __shfl_xor(s, m, 32);
  float mean = s * (1.f / PP);
  float vs = 0.f;
#pragma unroll
  for (int j = 0; j < 8; ++j) { float dt = v[j] - mean; vs += dt * dt; }
#pragma unroll
  for (int m = 1; m < 32; m <<= 1) vs += __shfl_xor(vs, m, 32);
  float rinv = rsqrtf(vs * (1.f / PP) + EPSF);
  __bf16* orow = out + (size_t)row * PP;
#pragma unroll
  for (int j = 0; j < 8; ++j) {
    int c = lane * 8 + j;
    orow[c] = (__bf16)((v[j] - mean) * rinv * g[c] + b[c]);
  }
}

// ---------------- flash attention + alpha*ao + residual ---------------------
// grid: (B*H, L/64), block: 128 (4 waves, 16 q-rows each). K/V tiles of 32 keys
// double-buffered in LDS; async global->LDS copies for tile i+1 overlap the
// WMMA + softmax work on tile i.
__global__ __launch_bounds__(128)
void k_attn(const __bf16* __restrict__ znbf, const __bf16* __restrict__ kvbf,
            const float* __restrict__ zres, const float* __restrict__ alpha_l,
            __bf16* __restrict__ aobf) {
  __shared__ __bf16 Ktile[2][32][64];
  __shared__ __bf16 Vtile[2][32][64];
  __shared__ __bf16 Pw[4][16][32];         // per-wave P round-trip (C-layout -> A-layout)

  const int lane = threadIdx.x & 31;
  const int wave = threadIdx.x >> 5;
  const int bb = blockIdx.x / HH;
  const int hh = blockIdx.x % HH;
  const int hcol = hh * HD;
  const int q0 = blockIdx.y * 64 + wave * 16;
  const int hf = lane >> 4;

  const __bf16* qbase = znbf + ((size_t)(bb * LL + q0)) * PP + hcol;
  v16bf aq0 = load_frag_a(qbase, PP, lane);        // channels 0..31
  v16bf aq1 = load_frag_a(qbase + 32, PP, lane);   // channels 32..63

  float mrow[8], lrow[8];
  v8f zero = {};
  v8f acc[4];
#pragma unroll
  for (int i = 0; i < 8; ++i) { mrow[i] = -1e30f; lrow[i] = 0.f; }
#pragma unroll
  for (int t = 0; t < 4; ++t) acc[t] = zero;

  const __bf16* ksrc0 = kvbf + (size_t)bb * LL * 512 + hcol;

  // stage one 32-key K/V tile pair into LDS buffer `buf` via async DMA
  auto stage = [&](int buf, int kb) {
    const __bf16* ksrc = ksrc0 + (size_t)kb * 512;
#pragma unroll
    for (int j = 0; j < 2; ++j) {
      int e = threadIdx.x * 2 + j;                   // 256 16B chunks per tile
      int key = e >> 3, c8 = e & 7;
      const __bf16* gk = ksrc + (size_t)key * 512 + c8 * 8;
      async_b128((uint32_t)(size_t)&Ktile[buf][key][c8 * 8], gk);
      async_b128((uint32_t)(size_t)&Vtile[buf][key][c8 * 8], gk + 256);
    }
  };

  stage(0, 0);
  int p = 0;
  for (int kb = 0; kb < LL; kb += 32) {
    wait_async0();                                   // own async ops complete
    __syncthreads();                                 // publish tile p to all waves
    if (kb + 32 < LL) stage(p ^ 1, kb + 32);         // overlap DMA with compute

    // S = q . K^T : two 16x16 key tiles, contraction hd=64 split as 2x32
    v8f s0 = zero, s1 = zero;
    s0 = wmma_bf16(aq0, load_frag_b_rows(&Ktile[p][0][0],  64, lane), s0);
    s0 = wmma_bf16(aq1, load_frag_b_rows(&Ktile[p][0][32], 64, lane), s0);
    s1 = wmma_bf16(aq0, load_frag_b_rows(&Ktile[p][16][0],  64, lane), s1);
    s1 = wmma_bf16(aq1, load_frag_b_rows(&Ktile[p][16][32], 64, lane), s1);

    // online softmax: row r = i + 8*half lives across the 16 lanes of this half
#pragma unroll
    for (int i = 0; i < 8; ++i) {
      float a0 = s0[i] * 0.125f, a1 = s1[i] * 0.125f;  // 1/sqrt(64)
      float mx = fmaxf(a0, a1);
#pragma unroll
      for (int m = 1; m < 16; m <<= 1) mx = fmaxf(mx, __shfl_xor(mx, m, 32));
      float mnew = fmaxf(mrow[i], mx);
      float corr = __expf(mrow[i] - mnew);
      float p0 = __expf(a0 - mnew), p1 = __expf(a1 - mnew);
      float rs = p0 + p1;
#pragma unroll
      for (int m = 1; m < 16; m <<= 1) rs += __shfl_xor(rs, m, 32);
      lrow[i] = lrow[i] * corr + rs;
      mrow[i] = mnew;
#pragma unroll
      for (int t = 0; t < 4; ++t) acc[t][i] *= corr;
      int r = i + 8 * hf, c = lane & 15;
      Pw[wave][r][c] = (__bf16)p0;
      Pw[wave][r][c + 16] = (__bf16)p1;
    }
    __builtin_amdgcn_wave_barrier();                 // LDS in-order within wave
    v16bf ap = load_frag_a(&Pw[wave][0][0], 32, lane);
#pragma unroll
    for (int t = 0; t < 4; ++t)
      acc[t] = wmma_bf16(ap, load_frag_b_strided(&Vtile[p][0][t * 16], 64, lane), acc[t]);
    __builtin_amdgcn_wave_barrier();
    p ^= 1;
  }

  const float alpha = alpha_l[0];
#pragma unroll
  for (int i = 0; i < 8; ++i) {
    float inv = 1.f / lrow[i];
    int ql = q0 + i + 8 * hf;
    size_t rowoff = ((size_t)(bb * LL + ql)) * PP + hcol + (lane & 15);
#pragma unroll
    for (int t = 0; t < 4; ++t) {
      float val = acc[t][i] * inv * alpha + zres[rowoff + t * 16];
      aobf[rowoff + t * 16] = (__bf16)val;
    }
  }
}

// ---------------- SRU recurrence (sequential in L, parallel over B*d) -------
__global__ __launch_bounds__(256)
void k_sru(const float* __restrict__ U, const float* __restrict__ hin,
           const float* __restrict__ wcl, const float* __restrict__ biasl,
           const float* __restrict__ c0l, float* __restrict__ hout,
           float* __restrict__ cfin) {
  int idx = blockIdx.x * 256 + threadIdx.x;   // 0..4095
  int bb = idx >> 10;
  int ch = idx & 1023;
  float vf = wcl[ch], vr = wcl[DD + ch];
  float bfv = biasl[ch], brv = biasl[DD + ch];
  float c = c0l[(size_t)bb * DD + ch];
  const float* Ub = U + (size_t)bb * LL * 3 * DD;
  const float* xb = hin + (size_t)bb * LL * DD;
  float* hb = hout + (size_t)bb * LL * DD;
  for (int t = 0; t < LL; ++t) {
    const float* u = Ub + (size_t)t * 3 * DD;
    __builtin_prefetch(u + 3 * DD + ch, 0, 1);
    float u0 = u[ch], u1 = u[DD + ch], u2 = u[2 * DD + ch];
    float x = xb[(size_t)t * DD + ch];
    float f = 1.f / (1.f + __expf(-(u1 + bfv + vf * c)));
    c = u0 + (c - u0) * f;
    float r = 1.f / (1.f + __expf(-(u2 + brv + vr * c)));
    hb[(size_t)t * DD + ch] = (c - x) * r + x;
  }
  cfin[(size_t)bb * DD + ch] = c;
}

// ---------------- host launch ----------------------------------------------
extern "C" void kernel_launch(void* const* d_in, const int* in_sizes, int n_in,
                              void* d_out, int out_size, void* d_ws, size_t ws_size,
                              hipStream_t stream) {
  (void)in_sizes; (void)n_in; (void)out_size; (void)ws_size;
  const float* x     = (const float*)d_in[0];
  const float* W1    = (const float*)d_in[1];
  const float* ln_g  = (const float*)d_in[2];
  const float* ln_b  = (const float*)d_in[3];
  const float* W2    = (const float*)d_in[4];
  const float* alpha = (const float*)d_in[5];
  const float* W3    = (const float*)d_in[6];
  const float* wc    = (const float*)d_in[7];
  const float* bias  = (const float*)d_in[8];
  const float* c0    = (const float*)d_in[9];

  float* out_h = (float*)d_out;
  float* out_c = out_h + (size_t)MTOT * DD;

  char* w = (char*)d_ws;
  size_t off = 0;
  auto alloc = [&](size_t bytes) -> void* {
    void* p = w + off;
    off = (off + bytes + 255) & ~(size_t)255;
    return p;
  };
  __bf16* hbf  = (__bf16*)alloc((size_t)MTOT * DD * 2);
  __bf16* w1t  = (__bf16*)alloc((size_t)DD * PP * 2);       // [PP][DD]
  __bf16* w2t  = (__bf16*)alloc((size_t)PP * 512 * 2);      // [512][PP]
  __bf16* w3t  = (__bf16*)alloc((size_t)PP * 3072 * 2);     // [3072][PP]
  float*  zbuf = (float*) alloc((size_t)MTOT * PP * 4);
  __bf16* znbf = (__bf16*)alloc((size_t)MTOT * PP * 2);
  __bf16* kvbf = (__bf16*)alloc((size_t)MTOT * 512 * 2);
  __bf16* aobf = (__bf16*)alloc((size_t)MTOT * PP * 2);
  float*  Ubuf = (float*) alloc((size_t)MTOT * 3 * DD * 4);
  float*  hbuf = (float*) alloc((size_t)MTOT * DD * 4);

  for (int l = 0; l < NLAYERS; ++l) {
    const float* hin = (l == 0) ? x : hbuf;
    k_cvt_bf16<<<dim3(2048), dim3(256), 0, stream>>>(hin, hbf, MTOT * DD);
    // weights: convert + transpose to [N][K] so GEMM B-fragments are contiguous
    k_cvt_tr_bf16<<<dim3(PP / 32, DD / 32), 256, 0, stream>>>(
        W1 + (size_t)l * DD * PP, w1t, DD, PP);
    k_cvt_tr_bf16<<<dim3(512 / 32, PP / 32), 256, 0, stream>>>(
        W2 + (size_t)l * PP * 512, w2t, PP, 512);
    k_cvt_tr_bf16<<<dim3(3072 / 32, PP / 32), 256, 0, stream>>>(
        W3 + (size_t)l * PP * 3072, w3t, PP, 3072);

    // z = h @ W1   (f32 out: LN input + attention residual)
    k_gemm_bf16<<<dim3(PP / 64, MTOT / 64), 128, 0, stream>>>(hbf, w1t, zbuf, nullptr,
                                                              MTOT, PP, DD);
    k_layernorm<<<dim3(MTOT / 8), 256, 0, stream>>>(zbuf, ln_g + (size_t)l * PP,
                                                    ln_b + (size_t)l * PP, znbf);
    // kv = zn @ W2 (bf16 out, only consumed by attention)
    k_gemm_bf16<<<dim3(512 / 64, MTOT / 64), 128, 0, stream>>>(znbf, w2t, nullptr, kvbf,
                                                               MTOT, 512, PP);
    k_attn<<<dim3(BB * HH, LL / 64), 128, 0, stream>>>(znbf, kvbf, zbuf, alpha + l, aobf);
    // U = ao @ W3  (f32 out for recurrence)
    k_gemm_bf16<<<dim3(3072 / 64, MTOT / 64), 128, 0, stream>>>(aobf, w3t, Ubuf, nullptr,
                                                                MTOT, 3072, PP);

    float* hout = (l == NLAYERS - 1) ? out_h : hbuf;
    k_sru<<<dim3(BB * DD / 256), 256, 0, stream>>>(Ubuf, hin, wc + (size_t)l * 2 * DD,
                                                   bias + (size_t)l * 2 * DD,
                                                   c0 + (size_t)l * BB * DD, hout,
                                                   out_c + (size_t)l * BB * DD);
  }
}
